// Decoder_74526272520762
// MI455X (gfx1250) — matrix operand order repaired and checked by reading, not compiled, not run on hardware
//
#include <hip/hip_runtime.h>
#include <hip/hip_bf16.h>
#include <math.h>

typedef _Float16 half_t;
typedef _Float16 v16h __attribute__((ext_vector_type(16)));
typedef __fp16   v2fp __attribute__((ext_vector_type(2)));
typedef float    v8f  __attribute__((ext_vector_type(8)));

#define SQRT_HALF 0.70710678118654752440f
#define LDW 21   // dword stride per LDS row/col (odd -> bank-conflict-free)

enum { EPI_PLAIN = 0, EPI_BIAS = 1, EPI_BIAS_RELU = 2, EPI_BIAS_SIGMOID = 3,
       EPI_SCORES = 4, EPI_OUTPROJ = 5 };

union FragU { v16h v; unsigned u[8]; };

__device__ __forceinline__ unsigned pk2(float a, float b) {
  union { v2fp h; unsigned u; } x;
  x.h = __builtin_amdgcn_cvt_pkrtz(a, b);   // v_cvt_pk_rtz_f16_f32
  return x.u;
}

__device__ __forceinline__ void pack_f4(unsigned* dst, const float4 f0,
                                        const float4 f1) {
  dst[0] = pk2(f0.x, f0.y);
  dst[1] = pk2(f0.z, f0.w);
  dst[2] = pk2(f1.x, f1.y);
  dst[3] = pk2(f1.z, f1.w);
}

struct GemmArgs {
  const float* A; long lda; long sAb;   // A: [M,K] row-major (+ batch stride)
  const float* B; long ldb; long sBb;   // B: [N,K] if BT else [K,N]
  float*       Y; long ldy; long sYb;   // Y: [M,N]
  int M, N, K;
  const float* bias;                    // [N] or null
  float alpha;                          // scale on accumulator
  const float* resid; long ldr; long sRb;
  float rscale;                         // scale after (acc*alpha+bias+resid)
  const int* prev_idx;                  // monotonic window mask (scores)
};

// ---------------------------------------------------------------------------
// f32-in / f16-WMMA / f32-out GEMM.  Block = 128 thr = 4 waves.
// Tile BM=64, BN=64, BK=32.  Wave (wr,wc) owns a 32x32 sub-tile -> 4 acc.
// Shapes are exact tile multiples (M%64==0, N%64==0, K%32==0) -> no guards.
// ---------------------------------------------------------------------------
template <bool BT>
__global__ __launch_bounds__(128)
void gemm_f16_wmma(GemmArgs g, int mode) {
  __shared__ unsigned Asu[64 * LDW];   // [row m][k pair]
  __shared__ unsigned Bsu[64 * LDW];   // [col n][k pair]

  const int tid  = threadIdx.x;
  const int wave = tid >> 5;
  const int lane = tid & 31;
  const int wr = wave >> 1, wc = wave & 1;
  const int h = lane >> 4, l16 = lane & 15;

  const int bz = blockIdx.z;
  const float* Ap = g.A + (long)bz * g.sAb;
  const float* Bp = g.B + (long)bz * g.sBb;
  float*       Yp = g.Y + (long)bz * g.sYb;

  const int m0 = blockIdx.y * 64;
  const int n0 = blockIdx.x * 64;

  v8f acc[2][2];
#pragma unroll
  for (int i = 0; i < 2; ++i)
#pragma unroll
    for (int j = 0; j < 2; ++j)
#pragma unroll
      for (int r = 0; r < 8; ++r) acc[i][j][r] = 0.0f;

  const int rA  = tid >> 1;            // 0..63
  const int kcA = (tid & 1) * 16;      // 0 or 16

  for (int k0 = 0; k0 < g.K; k0 += 32) {
    // ---- stage A tile 64x32 (fp32 -> packed f16 dwords) ----
    {
      const float* arow = Ap + (long)(m0 + rA) * g.lda + k0 + kcA;
      if (k0 + 32 < g.K) __builtin_prefetch(arow + 32, 0, 1);
      const float4 f0 = *(const float4*)(arow + 0);
      const float4 f1 = *(const float4*)(arow + 4);
      const float4 f2 = *(const float4*)(arow + 8);
      const float4 f3 = *(const float4*)(arow + 12);
      unsigned* dst = &Asu[rA * LDW + (kcA >> 1)];
      pack_f4(dst + 0, f0, f1);
      pack_f4(dst + 4, f2, f3);
    }
    // ---- stage B tile as [n][k] packed dwords ----
    if (BT) {
      const float* brow = Bp + (long)(n0 + rA) * g.ldb + k0 + kcA;
      const float4 f0 = *(const float4*)(brow + 0);
      const float4 f1 = *(const float4*)(brow + 4);
      const float4 f2 = *(const float4*)(brow + 8);
      const float4 f3 = *(const float4*)(brow + 12);
      unsigned* dst = &Bsu[rA * LDW + (kcA >> 1)];
      pack_f4(dst + 0, f0, f1);
      pack_f4(dst + 4, f2, f3);
    } else {
      const int n  = rA;                 // 0..63
      const int j0 = (tid & 1) * 8;      // k-pair base
      const float* bcol = Bp + (long)(k0 + 2 * j0) * g.ldb + n0 + n;
      unsigned* dst = &Bsu[n * LDW + j0];
#pragma unroll
      for (int j = 0; j < 8; ++j)
        dst[j] = pk2(bcol[(long)(2 * j) * g.ldb],
                     bcol[(long)(2 * j + 1) * g.ldb]);
    }
    __syncthreads();

    // ---- fragments: contiguous dword gathers, k = 16h + s in A and B ----
    FragU af[2], bf[2];
#pragma unroll
    for (int s = 0; s < 2; ++s) {
      const unsigned* arow = &Asu[(wr * 32 + s * 16 + l16) * LDW + 8 * h];
#pragma unroll
      for (int i = 0; i < 4; ++i) af[s].u[i] = arow[i];
#pragma unroll
      for (int i = 4; i < 8; ++i) af[s].u[i] = arow[i];
      const unsigned* bcol = &Bsu[(wc * 32 + s * 16 + l16) * LDW + 8 * h];
#pragma unroll
      for (int i = 0; i < 8; ++i) bf[s].u[i] = bcol[i];
    }
#pragma unroll
    for (int i = 0; i < 2; ++i)
#pragma unroll
      for (int j = 0; j < 2; ++j)
        acc[i][j] = __builtin_amdgcn_wmma_f32_16x16x32_f16(
            false, af[i].v, false, bf[j].v, (short)0, acc[i][j], false, false);
    __syncthreads();
  }

  // ---- epilogue ----
  const int pidx = (mode == EPI_SCORES && g.prev_idx) ? g.prev_idx[0] : -1;
  const float* Rp = g.resid ? (g.resid + (long)bz * g.sRb) : nullptr;
#pragma unroll
  for (int i = 0; i < 2; ++i)
#pragma unroll
    for (int j = 0; j < 2; ++j) {
      const int mb  = m0 + wr * 32 + i * 16;
      const int col = n0 + wc * 32 + j * 16 + l16;
      const float bv = (g.bias && mode != EPI_PLAIN && mode != EPI_SCORES)
                           ? g.bias[col] : 0.0f;
#pragma unroll
      for (int r = 0; r < 8; ++r) {
        const int row = mb + r + 8 * h;
        float v = acc[i][j][r] * g.alpha + bv;
        if (mode == EPI_BIAS_RELU) {
          v = fmaxf(v, 0.0f);
        } else if (mode == EPI_BIAS_SIGMOID) {
          v = 1.0f / (1.0f + __expf(-v));
        } else if (mode == EPI_SCORES) {
          if (pidx >= 0 && (col < pidx || col >= pidx + 3)) v = -INFINITY;
        } else if (mode == EPI_OUTPROJ) {
          v = (v + Rp[(long)row * g.ldr + col]) * g.rscale;
        }
        Yp[(long)row * g.ldy + col] = v;
      }
    }
}

// ---------------------------------------------------------------------------
// Causal dilated conv (K=5) + GLU + residual as 5 shifted WMMA GEMM taps.
// X,Y: [64*512, 256] fp32 (row = time within batch).  W: [512,256,5] fp32.
// Block = 128 thr = 4 waves; tile 32 rows x 32 GLU channels; each wave owns
// one 16x16 `a` accumulator and one 16x16 `g` accumulator.
// ---------------------------------------------------------------------------
__global__ __launch_bounds__(128)
void conv_glu_wmma(const float* __restrict__ X, const float* __restrict__ W,
                   const float* __restrict__ bias, float* __restrict__ Y,
                   int dil) {
  __shared__ unsigned Asu[32 * LDW];   // [row t][ic pair]
  __shared__ unsigned Bsu[64 * LDW];   // [col oc(a/g)][ic pair]

  const int tid  = threadIdx.x;
  const int wave = tid >> 5;
  const int lane = tid & 31;
  const int wr = wave >> 1, wc = wave & 1;
  const int h = lane >> 4, l16 = lane & 15;

  const int b   = blockIdx.z;
  const int t0  = blockIdx.y * 32;
  const int ac0 = blockIdx.x * 32;     // GLU output-channel base (0..255)

  v8f accA, accG;
#pragma unroll
  for (int r = 0; r < 8; ++r) { accA[r] = 0.0f; accG[r] = 0.0f; }

  // A staging coords: 32 rows x 16 dwords; 128 thr -> 4 dwords each
  const int rA  = tid >> 2;            // 0..31
  const int kcA = (tid & 3) * 8;       // float base 0/8/16/24
  // B staging coords: 64 cols x 16 dwords; 128 thr -> 8 dwords each
  const int nB  = tid >> 1;            // 0..63
  const int jB  = (tid & 1) * 8;       // dword base
  const int ocB = (nB < 32) ? (ac0 + nB) : (256 + ac0 + (nB - 32));

  for (int tap = 0; tap < 5; ++tap) {
    const int shift = (4 - tap) * dil;
    const float* wtap = W + (long)ocB * 1280 + tap;   // + ic*5
    for (int c0 = 0; c0 < 256; c0 += 32) {
      // ---- stage shifted A rows ----
      {
        const int t = t0 + rA - shift;
        unsigned* dst = &Asu[rA * LDW + (kcA >> 1)];
        if (t >= 0) {
          const float* xr = X + ((long)b * 512 + t) * 256 + c0 + kcA;
          const float4 f0 = *(const float4*)(xr + 0);
          const float4 f1 = *(const float4*)(xr + 4);
          pack_f4(dst, f0, f1);
        } else {
          dst[0] = 0u; dst[1] = 0u; dst[2] = 0u; dst[3] = 0u;
        }
      }
      // ---- stage weight tile [oc][ic pair], strided tap gather ----
      {
        const float* wp = wtap + (long)(c0 + 2 * jB) * 5;
        unsigned* dst = &Bsu[nB * LDW + jB];
#pragma unroll
        for (int j = 0; j < 8; ++j)
          dst[j] = pk2(wp[10 * j], wp[10 * j + 5]);
      }
      __syncthreads();

      FragU af, bfA, bfG;
      {
        const unsigned* arow = &Asu[(wr * 16 + l16) * LDW + 8 * h];
#pragma unroll
        for (int i = 0; i < 4; ++i) af.u[i] = arow[i];
#pragma unroll
        for (int i = 4; i < 8; ++i) af.u[i] = arow[i];
        const unsigned* bc0 = &Bsu[(wc * 16 + l16) * LDW + 8 * h];
        const unsigned* bc1 = &Bsu[(32 + wc * 16 + l16) * LDW + 8 * h];
#pragma unroll
        for (int i = 0; i < 8; ++i) { bfA.u[i] = bc0[i]; bfG.u[i] = bc1[i]; }
      }
      accA = __builtin_amdgcn_wmma_f32_16x16x32_f16(
          false, af.v, false, bfA.v, (short)0, accA, false, false);
      accG = __builtin_amdgcn_wmma_f32_16x16x32_f16(
          false, af.v, false, bfG.v, (short)0, accG, false, false);
      __syncthreads();
    }
  }

  // ---- GLU + residual epilogue ----
  const int acl = ac0 + wc * 16 + l16;
  const float ba = bias[acl];
  const float bg = bias[256 + acl];
#pragma unroll
  for (int r = 0; r < 8; ++r) {
    const int t   = t0 + wr * 16 + r + 8 * h;
    const long ro = ((long)b * 512 + t) * 256 + acl;
    const float a  = accA[r] + ba;
    const float gv = accG[r] + bg;
    const float y  = a * (1.0f / (1.0f + __expf(-gv)));
    Y[ro] = (y + X[ro]) * SQRT_HALF;
  }
}

// ---------------------------------------------------------------------------
// In-place row softmax over S columns; one block per row. Handles -inf mask.
// ---------------------------------------------------------------------------
__global__ __launch_bounds__(128)
void softmax_rows(float* __restrict__ P, int S) {
  __shared__ float red[4];
  float* p = P + (long)blockIdx.x * S;
  const int tid = threadIdx.x, wave = tid >> 5, lane = tid & 31;

  float m = -INFINITY;
  for (int i = tid; i < S; i += 128) m = fmaxf(m, p[i]);
  for (int o = 16; o; o >>= 1) m = fmaxf(m, __shfl_down(m, o));
  if (lane == 0) red[wave] = m;
  __syncthreads();
  if (tid == 0) {
    float mm = red[0];
    for (int i = 1; i < 4; ++i) mm = fmaxf(mm, red[i]);
    red[0] = mm;
  }
  __syncthreads();
  m = red[0];
  __syncthreads();

  float s = 0.0f;
  for (int i = tid; i < S; i += 128) {
    const float e = __expf(p[i] - m);
    p[i] = e;
    s += e;
  }
  for (int o = 16; o; o >>= 1) s += __shfl_down(s, o);
  if (lane == 0) red[wave] = s;
  __syncthreads();
  if (tid == 0) red[0] = red[0] + red[1] + red[2] + red[3];
  __syncthreads();
  const float inv = 1.0f / red[0];
  for (int i = tid; i < S; i += 128) p[i] *= inv;
}

// done = sigmoid(states @ w + b), matrix-vector -> plain VALU kernel
__global__ void done_kernel(const float* __restrict__ Xs,
                            const float* __restrict__ w,
                            const float* __restrict__ bsc,
                            float* __restrict__ out, int M) {
  const int i = blockIdx.x * blockDim.x + threadIdx.x;
  if (i >= M) return;
  const float* x = Xs + (long)i * 256;
  float s = 0.0f;
  for (int c = 0; c < 256; ++c) s += x[c] * w[c];
  s += bsc[0];
  out[i] = 1.0f / (1.0f + __expf(-s));
}

// ---------------------------------------------------------------------------
extern "C" void kernel_launch(void* const* d_in, const int* in_sizes, int n_in,
                              void* d_out, int out_size, void* d_ws,
                              size_t ws_size, hipStream_t stream) {
  (void)in_sizes; (void)n_in; (void)out_size; (void)ws_size;

  constexpr int B = 64, T = 512, S = 1024, Dd = 256, De = 512, INCH = 320;
  const int BT_ = B * T;    // 32768
  const int BS_ = B * S;    // 65536

  const float* inputs = (const float*)d_in[0];
  const float* keys   = (const float*)d_in[1];
  const float* values = (const float*)d_in[2];
  // params flatten order (jax pytree: dict keys sorted):
  // attn1{Wk,Wo,Wq,Wv,bk,bo,bq,bv}, attn4{...}, convs[(W,b)x4],
  // fc1_W, fc1_b, fc2_W, fc2_b, fc3_W, fc3_b, then prev_idx.
  const float* a1_Wk = (const float*)d_in[3];
  const float* a1_Wo = (const float*)d_in[4];
  const float* a1_Wq = (const float*)d_in[5];
  const float* a1_Wv = (const float*)d_in[6];
  const float* a1_bk = (const float*)d_in[7];
  const float* a1_bo = (const float*)d_in[8];
  const float* a1_bq = (const float*)d_in[9];
  const float* a1_bv = (const float*)d_in[10];
  const float* a4_Wk = (const float*)d_in[11];
  const float* a4_Wo = (const float*)d_in[12];
  const float* a4_Wq = (const float*)d_in[13];
  const float* a4_Wv = (const float*)d_in[14];
  const float* a4_bk = (const float*)d_in[15];
  const float* a4_bo = (const float*)d_in[16];
  const float* a4_bq = (const float*)d_in[17];
  const float* a4_bv = (const float*)d_in[18];
  const float* cW[4] = {(const float*)d_in[19], (const float*)d_in[21],
                        (const float*)d_in[23], (const float*)d_in[25]};
  const float* cB[4] = {(const float*)d_in[20], (const float*)d_in[22],
                        (const float*)d_in[24], (const float*)d_in[26]};
  const float* fc1_W = (const float*)d_in[27];
  const float* fc1_b = (const float*)d_in[28];
  const float* fc2_W = (const float*)d_in[29];
  const float* fc2_b = (const float*)d_in[30];
  const float* fc3_W = (const float*)d_in[31];
  const float* fc3_b = (const float*)d_in[32];
  const int* prev_idx = (const int*)d_in[33];

  // workspace carve-up (fp32)
  float* bufA = (float*)d_ws;                       // [BT_,256]
  float* bufB = bufA + (size_t)BT_ * Dd;            // [BT_,256]
  float* qc   = bufB + (size_t)BT_ * Dd;            // q, later ctx [BT_,256]
  float* kp   = qc   + (size_t)BT_ * Dd;            // [BS_,256]
  float* vp   = kp   + (size_t)BS_ * Dd;            // [BS_,256]

  // d_out carve-up
  float* out_o    = (float*)d_out;                  // [BT_,320]
  float* states_o = out_o + (size_t)BT_ * INCH;     // [BT_,256]
  float* done_o   = states_o + (size_t)BT_ * Dd;    // [BT_]
  float* align1   = done_o + BT_;                   // [B,T,S]
  float* align4   = align1 + (size_t)B * T * S;     // [B,T,S]

  auto gemm = [&](const float* A, long lda, long sAb, const float* Bm, long ldb,
                  long sBb, bool btrans, float* Y, long ldy, long sYb, int M,
                  int N, int K, int nb, const float* bias, float alpha,
                  const float* resid, long ldr, long sRb, float rscale,
                  int mode) {
    GemmArgs g{A, lda, sAb, Bm, ldb, sBb, Y, ldy, sYb, M, N, K,
               bias, alpha, resid, ldr, sRb, rscale, prev_idx};
    dim3 grid((N + 63) / 64, (M + 63) / 64, nb);
    if (btrans)
      gemm_f16_wmma<true><<<grid, 128, 0, stream>>>(g, mode);
    else
      gemm_f16_wmma<false><<<grid, 128, 0, stream>>>(g, mode);
  };

  auto attention = [&](const float* x /*[BT_,256]*/, const float* Wq,
                       const float* bq, const float* Wk, const float* bk,
                       const float* Wv, const float* bv, const float* Wo,
                       const float* bo, float* align, float* out) {
    // projections
    gemm(x, Dd, 0, Wq, Dd, 0, true, qc, Dd, 0, BT_, Dd, Dd, 1, bq, 1.f,
         nullptr, 0, 0, 1.f, EPI_BIAS);
    gemm(keys, De, 0, Wk, De, 0, true, kp, Dd, 0, BS_, Dd, De, 1, bk, 1.f,
         nullptr, 0, 0, 1.f, EPI_BIAS);
    gemm(values, De, 0, Wv, De, 0, true, vp, Dd, 0, BS_, Dd, De, 1, bv, 1.f,
         nullptr, 0, 0, 1.f, EPI_BIAS);
    // scores = q @ k^T  (batched over B), window mask via prev_idx
    gemm(qc, Dd, (long)T * Dd, kp, Dd, (long)S * Dd, true, align, S,
         (long)T * S, T, S, Dd, B, nullptr, 1.f, nullptr, 0, 0, 1.f,
         EPI_SCORES);
    softmax_rows<<<BT_, 128, 0, stream>>>(align, S);
    // ctx = attn @ v (batched); overwrite qc
    gemm(align, S, (long)T * S, vp, Dd, (long)S * Dd, false, qc, Dd,
         (long)T * Dd, T, Dd, S, B, nullptr, 1.f, nullptr, 0, 0, 1.f,
         EPI_PLAIN);
    // out = (sqrt(S)*ctx @ Wo^T + bo + residual) * sqrt(0.5)
    gemm(qc, Dd, 0, Wo, Dd, 0, true, out, Dd, 0, BT_, Dd, Dd, 1, bo, 32.0f,
         x, Dd, 0, SQRT_HALF, EPI_OUTPROJ);
  };

  const dim3 cgrid(Dd / 32, T / 32, B);

  // fc1 + relu
  gemm(inputs, INCH, 0, fc1_W, INCH, 0, true, bufA, Dd, 0, BT_, Dd, INCH, 1,
       fc1_b, 1.f, nullptr, 0, 0, 1.f, EPI_BIAS_RELU);
  // conv1 (dil 1)
  conv_glu_wmma<<<cgrid, 128, 0, stream>>>(bufA, cW[0], cB[0], bufB, 1);
  // attention 1 -> bufA
  attention(bufB, a1_Wq, a1_bq, a1_Wk, a1_bk, a1_Wv, a1_bv, a1_Wo, a1_bo,
            align1, bufA);
  // conv2..4 (dil 2,2,3)
  conv_glu_wmma<<<cgrid, 128, 0, stream>>>(bufA, cW[1], cB[1], bufB, 2);
  conv_glu_wmma<<<cgrid, 128, 0, stream>>>(bufB, cW[2], cB[2], bufA, 2);
  conv_glu_wmma<<<cgrid, 128, 0, stream>>>(bufA, cW[3], cB[3], bufB, 3);
  // attention 4 -> decoder_states (written straight into d_out)
  attention(bufB, a4_Wq, a4_bq, a4_Wk, a4_bk, a4_Wv, a4_bv, a4_Wo, a4_bo,
            align4, states_o);
  // out = sigmoid(states @ fc2_W^T + b)
  gemm(states_o, Dd, 0, fc2_W, Dd, 0, true, out_o, INCH, 0, BT_, INCH, Dd, 1,
       fc2_b, 1.f, nullptr, 0, 0, 1.f, EPI_BIAS_SIGMOID);
  // done = sigmoid(states @ fc3_W^T + b)
  done_kernel<<<(BT_ + 255) / 256, 256, 0, stream>>>(states_o, fc3_W, fc3_b,
                                                     done_o, BT_);
}
